// SpectralRewiringLayer_34024730919237
// MI455X (gfx1250) — matrix operand order repaired and checked by reading, not compiled, not run on hardware
//
#include <hip/hip_runtime.h>
#include <hip/hip_bf16.h>

typedef _Float16 v16h __attribute__((ext_vector_type(16)));
typedef _Float16 v8h  __attribute__((ext_vector_type(8)));
typedef _Float16 v4h  __attribute__((ext_vector_type(4)));
typedef _Float16 v2h  __attribute__((ext_vector_type(2)));
typedef float    v8f  __attribute__((ext_vector_type(8)));

#define NT  4   // N tiles of 16 (H=64)
#define KT1 5   // layer1 K tiles of 32 (130 padded to 160)
#define KT2 2   // layer2 K tiles of 32 (64)

// ---------------------------------------------------------------------------
// Pre-swizzle W1 (130x64, zero-padded to 160x64) and W2 (64x64) from fp32 into
// the dense-B per-lane WMMA register image, f16:
//   lane l holds B[k][n] for n = l&15, k = ktile*32 + (l<16 ? 0 : 16) + i,
//   i = 0..15, stored contiguously (32 bytes per lane-slot).
// ---------------------------------------------------------------------------
__global__ void prep_weights_kernel(const float* __restrict__ W1,
                                    const float* __restrict__ W2,
                                    _Float16* __restrict__ w1img,
                                    _Float16* __restrict__ w2img) {
  int t = blockIdx.x * blockDim.x + threadIdx.x;
  if (t < NT * KT1 * 32) {
    int lane = t & 31;
    int kt   = (t >> 5) % KT1;
    int nt   = t / (32 * KT1);
    int n    = nt * 16 + (lane & 15);
    int kst  = kt * 32 + ((lane < 16) ? 0 : 16);
#pragma unroll
    for (int i = 0; i < 16; ++i) {
      int k = kst + i;
      float v = (k < 130) ? W1[k * 64 + n] : 0.0f;
      w1img[t * 16 + i] = (_Float16)v;
    }
  } else if (t < NT * KT1 * 32 + NT * KT2 * 32) {
    int t2   = t - NT * KT1 * 32;
    int lane = t2 & 31;
    int kt   = (t2 >> 5) % KT2;
    int nt   = t2 / (32 * KT2);
    int n    = nt * 16 + (lane & 15);
    int kst  = kt * 32 + ((lane < 16) ? 0 : 16);
#pragma unroll
    for (int i = 0; i < 16; ++i)
      w2img[t2 * 16 + i] = (_Float16)W2[(kst + i) * 64 + n];
  }
}

// ---------------------------------------------------------------------------
// Main kernel: one wave32 per 32-edge tile (two M=16 subtiles sharing every
// B-operand load); 4 waves per block.
//   L1: 2 x ([16x160]f16 x [160x64]f16 -> f32)  (40 WMMA)
//   L2: 2 x ([16x64]     x [64x64]     -> f32)  (16 WMMA)
//   L3: per-lane 64-dot on VALU (one output row per lane)
// ---------------------------------------------------------------------------
__global__ __launch_bounds__(128) void edge_mlp_kernel(
    const float* __restrict__ emb, const float* __restrict__ fied,
    const _Float16* __restrict__ w1img, const _Float16* __restrict__ w2img,
    const float* __restrict__ b1, const float* __restrict__ b2,
    const float* __restrict__ W3, const float* __restrict__ b3,
    const int* __restrict__ eidx, float* __restrict__ out, int E) {
  __shared__ __align__(32) _Float16 sfeat[4][32][160];  // per-wave A staging (40 KB)
  __shared__ __align__(32) _Float16 sh[4][32][64];      // per-wave h1/h2    (16 KB)

  const int lane  = threadIdx.x & 31;
  const int wid   = threadIdx.x >> 5;
  const int m     = lane & 15;            // row within an M=16 subtile
  const int baseA = (lane < 16) ? 0 : 8;  // A-operand half-wave K base (and D row base)
  const int tile  = blockIdx.x * 4 + wid; // 32-edge tile index

  const long long tbase = (long long)tile * 32;
  const long long e  = tbase + lane;      // this lane's own edge (one row per lane)
  const bool valid   = e < (long long)E;
  const long long ec = valid ? e : (long long)(E - 1);
  const int srcn = eidx[ec];
  const int dstn = eidx[(long long)E + ec];

  // ---- gather: each lane builds its own feature row [160] in f16 ----
  {
    _Float16* frow = &sfeat[wid][lane][0];
    const float4* srow = (const float4*)(emb + (long long)srcn * 64);
    const float4* drow = (const float4*)(emb + (long long)dstn * 64);
#pragma unroll
    for (int j = 0; j < 16; ++j) {
      float4 v = srow[j];
      v4h h = {(_Float16)v.x, (_Float16)v.y, (_Float16)v.z, (_Float16)v.w};
      *(v4h*)(frow + j * 4) = h;
    }
#pragma unroll
    for (int j = 0; j < 16; ++j) {
      float4 v = drow[j];
      v4h h = {(_Float16)v.x, (_Float16)v.y, (_Float16)v.z, (_Float16)v.w};
      *(v4h*)(frow + 64 + j * 4) = h;
    }
    frow[128] = (_Float16)fied[srcn];
    frow[129] = (_Float16)fied[dstn];
#pragma unroll
    for (int j = 0; j < 15; ++j) {  // zero pad K = 130..159
      v2h z = {(_Float16)0.0f, (_Float16)0.0f};
      *(v2h*)(frow + 130 + 2 * j) = z;
    }
  }
  // LDS ops within a wave are in-order; compiler inserts dscnt waits on loads.

  const _Float16* arow0 = &sfeat[wid][m][0];       // subtile 0: rows 0..15
  const _Float16* arow1 = &sfeat[wid][16 + m][0];  // subtile 1: rows 16..31

  // ---- layer 1: 5 K-tiles x 4 N-tiles x 2 M-subtiles ----
  const v16h* B1 = (const v16h*)w1img;
  v8f c0[NT] = {}, c1[NT] = {};
#pragma unroll
  for (int kt = 0; kt < KT1; ++kt) {
    v8h a0lo = *(const v8h*)(arow0 + kt * 32 + 2 * baseA);
    v8h a0hi = *(const v8h*)(arow0 + kt * 32 + 8 + 2 * baseA);
    v8h a1lo = *(const v8h*)(arow1 + kt * 32 + 2 * baseA);
    v8h a1hi = *(const v8h*)(arow1 + kt * 32 + 8 + 2 * baseA);
    v16h a0 = __builtin_shufflevector(a0lo, a0hi, 0, 1, 2, 3, 4, 5, 6, 7,
                                      8, 9, 10, 11, 12, 13, 14, 15);
    v16h a1 = __builtin_shufflevector(a1lo, a1hi, 0, 1, 2, 3, 4, 5, 6, 7,
                                      8, 9, 10, 11, 12, 13, 14, 15);
#pragma unroll
    for (int nt = 0; nt < NT; ++nt) {
      v16h b = B1[(nt * KT1 + kt) * 32 + lane];  // one B load feeds two WMMAs
      c0[nt] = __builtin_amdgcn_wmma_f32_16x16x32_f16(false, a0, false, b, (short)0, c0[nt], false, false);
      c1[nt] = __builtin_amdgcn_wmma_f32_16x16x32_f16(false, a1, false, b, (short)0, c1[nt], false, false);
    }
  }

  // ---- h1 = relu(c + b1) -> LDS (D layout: VGPR r -> row baseA+r, col = nt*16 + lane&15)
  {
    const int col = lane & 15;
#pragma unroll
    for (int nt = 0; nt < NT; ++nt) {
      float bb = b1[nt * 16 + col];
#pragma unroll
      for (int r = 0; r < 8; ++r) {
        float v0 = c0[nt][r] + bb;
        float v1 = c1[nt][r] + bb;
        v0 = v0 > 0.0f ? v0 : 0.0f;
        v1 = v1 > 0.0f ? v1 : 0.0f;
        sh[wid][baseA + r][nt * 16 + col]      = (_Float16)v0;
        sh[wid][16 + baseA + r][nt * 16 + col] = (_Float16)v1;
      }
    }
  }

  // ---- layer 2: 2 K-tiles x 4 N-tiles x 2 M-subtiles ----
  const v16h* B2 = (const v16h*)w2img;
  const _Float16* hrow0 = &sh[wid][m][0];
  const _Float16* hrow1 = &sh[wid][16 + m][0];
  v8f d0[NT] = {}, d1[NT] = {};
#pragma unroll
  for (int kt = 0; kt < KT2; ++kt) {
    v8h a0lo = *(const v8h*)(hrow0 + kt * 32 + 2 * baseA);
    v8h a0hi = *(const v8h*)(hrow0 + kt * 32 + 8 + 2 * baseA);
    v8h a1lo = *(const v8h*)(hrow1 + kt * 32 + 2 * baseA);
    v8h a1hi = *(const v8h*)(hrow1 + kt * 32 + 8 + 2 * baseA);
    v16h a0 = __builtin_shufflevector(a0lo, a0hi, 0, 1, 2, 3, 4, 5, 6, 7,
                                      8, 9, 10, 11, 12, 13, 14, 15);
    v16h a1 = __builtin_shufflevector(a1lo, a1hi, 0, 1, 2, 3, 4, 5, 6, 7,
                                      8, 9, 10, 11, 12, 13, 14, 15);
#pragma unroll
    for (int nt = 0; nt < NT; ++nt) {
      v16h b = B2[(nt * KT2 + kt) * 32 + lane];
      d0[nt] = __builtin_amdgcn_wmma_f32_16x16x32_f16(false, a0, false, b, (short)0, d0[nt], false, false);
      d1[nt] = __builtin_amdgcn_wmma_f32_16x16x32_f16(false, a1, false, b, (short)0, d1[nt], false, false);
    }
  }

  // ---- h2 = relu(d + b2) -> LDS (overwrite h1) ----
  {
    const int col = lane & 15;
#pragma unroll
    for (int nt = 0; nt < NT; ++nt) {
      float bb = b2[nt * 16 + col];
#pragma unroll
      for (int r = 0; r < 8; ++r) {
        float v0 = d0[nt][r] + bb;
        float v1 = d1[nt][r] + bb;
        v0 = v0 > 0.0f ? v0 : 0.0f;
        v1 = v1 > 0.0f ? v1 : 0.0f;
        sh[wid][baseA + r][nt * 16 + col]      = (_Float16)v0;
        sh[wid][16 + baseA + r][nt * 16 + col] = (_Float16)v1;
      }
    }
  }

  // ---- layer 3: one output row per lane: score = h2[lane] . W3 + b3 ----
  {
    const _Float16* hr = &sh[wid][lane][0];
    float s = 0.0f;
#pragma unroll
    for (int k = 0; k < 64; ++k)
      s += (float)hr[k] * W3[k];
    if (valid) out[e] = s + b3[0];
  }
}

// ---------------------------------------------------------------------------
extern "C" void kernel_launch(void* const* d_in, const int* in_sizes, int n_in,
                              void* d_out, int out_size, void* d_ws, size_t ws_size,
                              hipStream_t stream) {
  const float* emb  = (const float*)d_in[0];  // node_embeddings [N,64]
  const float* fied = (const float*)d_in[1];  // fiedler [N]
  const float* W1   = (const float*)d_in[2];  // [130,64]
  const float* b1   = (const float*)d_in[3];  // [64]
  const float* W2   = (const float*)d_in[4];  // [64,64]
  const float* b2   = (const float*)d_in[5];  // [64]
  const float* W3   = (const float*)d_in[6];  // [64,1]
  const float* b3   = (const float*)d_in[7];  // [1]
  const int*   eidx = (const int*)d_in[8];    // edge_index [2,E]
  float* out = (float*)d_out;

  const int E = in_sizes[8] / 2;

  _Float16* w1img = (_Float16*)d_ws;                    // 4*5*32*16 halfs = 20480 B
  _Float16* w2img = w1img + NT * KT1 * 32 * 16;         // 4*2*32*16 halfs =  8192 B

  // one-time (per launch) weight re-layout into WMMA B register image
  prep_weights_kernel<<<4, 256, 0, stream>>>(W1, W2, w1img, w2img);

  const int tiles  = (E + 31) / 32;     // 32 edges per wave
  const int blocks = (tiles + 3) / 4;   // 4 waves per block
  edge_mlp_kernel<<<blocks, 128, 0, stream>>>(emb, fied, w1img, w2img,
                                              b1, b2, W3, b3, eidx, out, E);
}
